// LocalWindowMaskedHead_790273982532
// MI455X (gfx1250) — matrix operand order repaired and checked by reading, not compiled, not run on hardware
//
#include <hip/hip_runtime.h>
#include <hip/hip_bf16.h>
#include <math.h>

typedef _Float16 half_t;
typedef __attribute__((ext_vector_type(16))) _Float16 v16h;
typedef __attribute__((ext_vector_type(8)))  _Float16 v8h;
typedef __attribute__((ext_vector_type(8)))  float    v8f;

#define TT 32   // sequence tile (BLOCK_SIZE)
#define CC 64   // model dim
#define HH 64   // head size
#define WIN 8   // local window

// ---- WMMA fragment helpers (layouts per CDNA5 ISA 7.12.2, wave32) ----

// A fragment: 16x32 f16 tile from row-major [rows][ld] matrix (ld in halves).
// lane<16: row=row0+lane, K=k0+{0..7},{16..23}; lane>=16: K=k0+{8..15},{24..31}
__device__ __forceinline__ v16h load_frag_a(const half_t* src, int ld,
                                            int row0, int k0, int lane) {
    int r  = row0 + (lane & 15);
    int kb = k0 + ((lane >> 4) << 3);
    const half_t* p = src + r * ld + kb;
    v8h lo = *(const v8h*)p;
    v8h hi = *(const v8h*)(p + 16);
    v16h a;
#pragma unroll
    for (int i = 0; i < 8; ++i) { a[i] = lo[i]; a[8 + i] = hi[i]; }
    return a;
}

// B fragment: 32x16 tile, source stored N-major: src[n][k], row stride ld (halves).
// lane<16: col n=n0+lane, K=k0+{0..7},{16..23} ; lane>=16: col n=n0+lane-16, K=k0+{8..15},{24..31}
// Two 16B reads per lane, at kb and kb+16.
__device__ __forceinline__ v16h load_frag_b_nmajor(const half_t* src, int ld,
                                                   int n0, int k0, int lane) {
    int n  = n0 + (lane & 15);
    int kb = k0 + ((lane >> 4) << 3);
    const half_t* p = src + n * ld + kb;
    v8h lo = *(const v8h*)p;
    v8h hi = *(const v8h*)(p + 16);
    v16h b;
#pragma unroll
    for (int i = 0; i < 8; ++i) { b[i] = lo[i]; b[8 + i] = hi[i]; }
    return b;
}

// Store f32 C/D tile to f16 row-major LDS.
// VGPR r: lanes 0-15 -> M=m0+r, lanes 16-31 -> M=m0+8+r ; N = n0+(lane&15)
__device__ __forceinline__ void store_tile_f16(half_t* dst, int ld, int m0, int n0,
                                               int lane, v8f c) {
    int col = n0 + (lane & 15);
    int rb  = m0 + ((lane >> 4) << 3);
#pragma unroll
    for (int r = 0; r < 8; ++r) dst[(rb + r) * ld + col] = (half_t)c[r];
}

// ---- prep kernel: Wq/Wk/Wv (f32, [C][H] k-major) -> f16 W^T[3][H][C] in ws ----
__global__ __launch_bounds__(256)
void lwmh_wprep_kernel(const float* __restrict__ wq, const float* __restrict__ wk,
                       const float* __restrict__ wv, half_t* __restrict__ wT) {
    int tid = blockIdx.x * 256 + threadIdx.x;
    if (tid >= 3 * CC * HH) return;
    int m = tid >> 12;
    int r = tid & 4095;
    int k = r >> 6;          // model-dim index
    int n = r & 63;          // head-dim index
    const float* w = (m == 0) ? wq : (m == 1) ? wk : wv;
    wT[m * 4096 + n * CC + k] = (half_t)w[k * HH + n];
}

__global__ __launch_bounds__(128)
void lwmh_attn_kernel(const float* __restrict__ xg,
                      const half_t* __restrict__ wTg,
                      float* __restrict__ outg,
                      float* __restrict__ attng) {
    // xstage (f32, 8KB) is dead after stage 0; att/atth live from stage 2 on.
    __shared__ __align__(16) unsigned char smem_u[TT * CC * 4];   // 8 KB union
    __shared__ half_t xh[TT][CC];        // 4 KB  x  (f16, row-major [t][c])
    __shared__ half_t wh[3][HH][CC];     // 24 KB W^T (f16, n-major [n][k])
    __shared__ half_t qh[TT][HH];        // 4 KB
    __shared__ half_t kh[TT][HH];        // 4 KB
    __shared__ half_t vT[HH][TT];        // 4 KB  v^T (n-major [h][s])

    float  (*att)[TT]  = (float (*)[TT])smem_u;                 // 4 KB
    half_t (*atth)[TT] = (half_t (*)[TT])(smem_u + 4096);       // 2 KB
    float* xstage      = (float*)smem_u;                        // 8 KB (stage 0 only)

    const int tid  = threadIdx.x;
    const int lane = tid & 31;
    const int wv   = tid >> 5;           // wave id 0..3
    const int bidx = blockIdx.x;

    // ---- Stage 0a: async-copy x tile (f32, 8KB) global -> LDS staging ----
    {
        uint32_t ldsbase = (uint32_t)(uintptr_t)xstage;
        const float* gx = xg + (size_t)bidx * (TT * CC);
#pragma unroll
        for (int i = 0; i < 4; ++i) {    // 16 chunks of 512B, 4 per wave
            uint32_t off = (uint32_t)(wv * 4 + i) * 512u + (uint32_t)lane * 16u;
            asm volatile("global_load_async_to_lds_b128 %0, %1, %2"
                         :: "v"(ldsbase + off), "v"(off), "s"(gx) : "memory");
        }
        // meanwhile: copy pre-transposed f16 weights ws -> LDS (plain b128)
        const uint4* wsrc = (const uint4*)wTg;
        uint4*       wdst = (uint4*)&wh[0][0][0];
#pragma unroll
        for (int i = 0; i < 12; ++i)     // 1536 x 16B / 128 threads
            wdst[tid + i * 128] = wsrc[tid + i * 128];
        asm volatile("s_wait_asynccnt 0x0" ::: "memory");
    }
    __syncthreads();

    // ---- Stage 0b: convert x f32 -> f16 ----
    {
        half_t* xd = &xh[0][0];
#pragma unroll
        for (int i = 0; i < 4; ++i) {
            int idx = tid + i * 128;
            float4 f = ((const float4*)xstage)[idx];
            v8h h4;  // unused; the 4 halves are stored singly
            xd[idx * 4 + 0] = (half_t)f.x;
            xd[idx * 4 + 1] = (half_t)f.y;
            xd[idx * 4 + 2] = (half_t)f.z;
            xd[idx * 4 + 3] = (half_t)f.w;
            (void)h4;
        }
    }
    __syncthreads();

    // ---- Stage 1: q,k,v = x @ W (24 tiles of 16x16, 6 per wave, K=64) ----
#pragma unroll
    for (int t6 = 0; t6 < 6; ++t6) {
        int t  = wv * 6 + t6;            // 0..23
        int m  = t >> 3;                 // 0:q 1:k 2:v
        int mt = (t & 7) >> 2;
        int nt = t & 3;
        v8f acc = {};
#pragma unroll
        for (int kk = 0; kk < CC; kk += 32) {
            v16h a = load_frag_a(&xh[0][0], CC, mt * 16, kk, lane);
            v16h b = load_frag_b_nmajor(&wh[m][0][0], CC, nt * 16, kk, lane);
            acc = __builtin_amdgcn_wmma_f32_16x16x32_f16(false, a, false, b,
                                                         (short)0, acc, false, false);
        }
        if (m == 2) {
            // store v transposed: vT[col][rb..rb+7] is contiguous -> one b128
            int col = nt * 16 + (lane & 15);
            int rb  = mt * 16 + ((lane >> 4) << 3);
            v8h hv;
#pragma unroll
            for (int r = 0; r < 8; ++r) hv[r] = (half_t)acc[r];
            *(v8h*)&vT[col][rb] = hv;
        } else {
            store_tile_f16(m == 0 ? &qh[0][0] : &kh[0][0], HH, mt * 16, nt * 16, lane, acc);
        }
    }
    __syncthreads();

    // ---- Stage 2: scores = q @ k^T * scale (4 tiles, 1 per wave, K=64) ----
    {
        int mt = wv >> 1, nt = wv & 1;
        v8f acc = {};
#pragma unroll
        for (int kk = 0; kk < HH; kk += 32) {
            v16h a = load_frag_a(&qh[0][0], HH, mt * 16, kk, lane);
            v16h b = load_frag_b_nmajor(&kh[0][0], HH, nt * 16, kk, lane);
            acc = __builtin_amdgcn_wmma_f32_16x16x32_f16(false, a, false, b,
                                                         (short)0, acc, false, false);
        }
        int col = nt * 16 + (lane & 15);
        int rb  = mt * 16 + ((lane >> 4) << 3);
#pragma unroll
        for (int r = 0; r < 8; ++r) att[rb + r][col] = acc[r] * 0.125f;
    }
    __syncthreads();

    // ---- Stage 3: masked local softmax, one row per thread ----
    if (tid < TT) {
        int i   = tid;
        int jlo = (i - (WIN - 1) > 0) ? i - (WIN - 1) : 0;
        float mx = -3.4e38f;
        for (int j = jlo; j <= i; ++j) mx = fmaxf(mx, att[i][j]);
        float sum = 0.f;
        for (int j = jlo; j <= i; ++j) {
            float e = __expf(att[i][j] - mx);
            att[i][j] = e;
            sum += e;
        }
        float inv = 1.0f / sum;
#pragma unroll
        for (int j = 0; j < TT; ++j) {
            float w = (j >= jlo && j <= i) ? att[i][j] * inv : 0.f;
            att[i][j]  = w;
            atth[i][j] = (half_t)w;
        }
    }
    __syncthreads();

    // ---- Stage 4: write attention weights (coalesced float4) ----
    {
        float4* ad = (float4*)(attng + (size_t)bidx * (TT * TT));
        const float4* as = (const float4*)&att[0][0];
#pragma unroll
        for (int i = 0; i < 2; ++i) ad[tid + i * 128] = as[tid + i * 128];
    }

    // ---- Stage 5: out = attn @ v (8 tiles, 2 per wave, K=32) ----
    float* outb = outg + (size_t)bidx * (TT * HH);
#pragma unroll
    for (int t2 = 0; t2 < 2; ++t2) {
        int t  = wv * 2 + t2;            // 0..7
        int mt = t >> 2, nt = t & 3;
        v16h a = load_frag_a(&atth[0][0], TT, mt * 16, 0, lane);
        v16h b = load_frag_b_nmajor(&vT[0][0], TT, nt * 16, 0, lane);
        v8f acc = {};
        acc = __builtin_amdgcn_wmma_f32_16x16x32_f16(false, a, false, b,
                                                     (short)0, acc, false, false);
        int col = nt * 16 + (lane & 15);
        int rb  = mt * 16 + ((lane >> 4) << 3);
#pragma unroll
        for (int r = 0; r < 8; ++r) outb[(rb + r) * HH + col] = acc[r];
    }
}

extern "C" void kernel_launch(void* const* d_in, const int* in_sizes, int n_in,
                              void* d_out, int out_size, void* d_ws, size_t ws_size,
                              hipStream_t stream) {
    const float* x  = (const float*)d_in[0];
    const float* Wq = (const float*)d_in[1];
    const float* Wk = (const float*)d_in[2];
    const float* Wv = (const float*)d_in[3];
    const int nb = in_sizes[0] / (TT * CC);
    float*  out  = (float*)d_out;                       // [nb, 32, 64]
    float*  attn = out + (size_t)nb * (TT * HH);        // [nb, 32, 32]
    half_t* wT   = (half_t*)d_ws;                       // [3][64][64] f16, n-major

    lwmh_wprep_kernel<<<dim3((3 * CC * HH + 255) / 256), dim3(256), 0, stream>>>(Wq, Wk, Wv, wT);
    lwmh_attn_kernel<<<dim3(nb), dim3(128), 0, stream>>>(x, wT, out, attn);
}
